// XlnetAttention_23433341567288
// MI455X (gfx1250) — matrix operand order repaired and checked by reading, not compiled, not run on hardware
//
#include <hip/hip_runtime.h>
#include <math.h>

// ---------------- problem constants ----------------
constexpr int Q_LEN = 512;
constexpr int B_SZ  = 8;
constexpr int HIDN  = 1024;
constexpr int N_H   = 16;
constexpr int H_D   = 64;
constexpr int P_LEN = 1024;
constexpr float SCALE_F = 0.125f;   // 1/sqrt(64)
constexpr float BIG_F   = 1.0e30f;

constexpr int MROWS = Q_LEN * B_SZ;   // 4096 (i*8+b)
constexpr int PROWS = P_LEN * B_SZ;   // 8192

// ---------------- WMMA vector types ----------------
typedef __attribute__((ext_vector_type(16))) __bf16 v16bf;
typedef __attribute__((ext_vector_type(8)))  __bf16 v8bf;
typedef __attribute__((ext_vector_type(8)))  float  v8f;

__device__ inline v8f wmma_bf16(v16bf a, v16bf b, v8f c) {
  return __builtin_amdgcn_wmma_f32_16x16x32_bf16(
      false, a, false, b, (short)0, c, false, false);
}

// A-fragment (16x32 bf16) from row-major [row][k], stride ld elements.
// Lane m(&15), half h(>>4): a[e] = A[row0+m][k0+16h+e]
__device__ inline v16bf load_a_frag(const __bf16* __restrict__ base, int ld,
                                    int row0, int k0, int lane) {
  int m = lane & 15, h = lane >> 4;
  const __bf16* p = base + (size_t)(row0 + m) * ld + k0 + 16 * h;
  v8bf lo = *(const v8bf*)p;
  v8bf hi = *(const v8bf*)(p + 8);
  v16bf a;
#pragma unroll
  for (int e = 0; e < 8; ++e) { a[e] = lo[e]; a[e + 8] = hi[e]; }
  return a;
}

// B-fragment (32x16 bf16) from B^T row-major [n][k], stride ld.
// Lane n(&15), half h(>>4): b[e] = Bt[n0+n][k0+16h+e]  (one 32B load)
__device__ inline v16bf load_b_frag(const __bf16* __restrict__ bt, int ld,
                                    int n0, int k0, int lane) {
  int n = lane & 15, h = lane >> 4;
  return *(const v16bf*)(bt + (size_t)(n0 + n) * ld + k0 + 16 * h);
}

// ---------------- prep kernels ----------------
__global__ __launch_bounds__(256) void cvt_bf16_kernel(
    const float* __restrict__ in, __bf16* __restrict__ out, size_t n) {
  size_t i = (size_t)blockIdx.x * blockDim.x + threadIdx.x;
  if (i < n) out[i] = (__bf16)in[i];
}

// weights [h][col] f32 -> W^T [col][h] bf16   (1024x1024)
__global__ __launch_bounds__(256) void wtrans_kernel(
    const float* __restrict__ in, __bf16* __restrict__ out) {
  int col = blockIdx.x * 16 + (threadIdx.x & 15);
  int h   = blockIdx.y * 16 + (threadIdx.x >> 4);
  out[(size_t)col * HIDN + h] = (__bf16)in[(size_t)h * HIDN + col];
}

// ---------------- generic bf16 WMMA GEMM --------------------------------
// One 16(m) x 64(n) tile per wave: the A-fragment is reused across 4
// B-fragments -> 4 WMMAs per A load (A traffic cut 4x vs 16x16 tiling).
// C[m,n] = sum_k A[m,k] * Bt[n,k].  Epilogue modes scatter into the
// attention layouts.  Grid must supply exactly (M/16)*(N/64) waves.
__global__ __launch_bounds__(256) void gemm_kernel(
    const __bf16* __restrict__ A, const __bf16* __restrict__ Bt,
    int tiles_n4, int K, int mode,
    const float* __restrict__ bias0, const float* __restrict__ bias1,
    __bf16* __restrict__ out0, __bf16* __restrict__ out1,
    float* __restrict__ outf) {
  int wave = threadIdx.x >> 5, lane = threadIdx.x & 31;
  int li = lane & 15, lh = lane >> 4;
  int t  = blockIdx.x * 8 + wave;
  int tm = t / tiles_n4, tn = t % tiles_n4;
  int row0 = tm * 16, col0 = tn * 64;
  v8f c0 = {}, c1 = {}, c2 = {}, c3 = {};
  for (int k0 = 0; k0 < K; k0 += 32) {
    v16bf a  = load_a_frag(A, K, row0, k0, lane);
    v16bf b0 = load_b_frag(Bt, K, col0,      k0, lane);
    v16bf b1 = load_b_frag(Bt, K, col0 + 16, k0, lane);
    v16bf b2 = load_b_frag(Bt, K, col0 + 32, k0, lane);
    v16bf b3 = load_b_frag(Bt, K, col0 + 48, k0, lane);
    if (k0 + 32 < K) {   // stream-ahead hint -> global_prefetch_b8
      __builtin_prefetch(A  + (size_t)(row0 + li) * K + k0 + 32, 0, 3);
      __builtin_prefetch(Bt + (size_t)(col0 + li) * K + k0 + 32, 0, 3);
    }
    c0 = wmma_bf16(a, b0, c0);
    c1 = wmma_bf16(a, b1, c1);
    c2 = wmma_bf16(a, b2, c2);
    c3 = wmma_bf16(a, b3, c3);
  }
  int N = tiles_n4 * 64;
#pragma unroll
  for (int sub = 0; sub < 4; ++sub) {
    v8f c = (sub == 0) ? c0 : (sub == 1) ? c1 : (sub == 2) ? c2 : c3;
    int col = col0 + sub * 16 + li;
    if (mode == 0) {                       // Q + two rel biases -> qw, qr
      float b0v = bias0[col], b1v = bias1[col];
#pragma unroll
      for (int r = 0; r < 8; ++r) {
        int row = row0 + 8 * lh + r;
        out0[(size_t)row * N + col] = (__bf16)(c[r] + b0v);
        out1[(size_t)row * N + col] = (__bf16)(c[r] + b1v);
      }
    } else if (mode == 1) {                // K -> [b][n][j][d]
      int nh = col >> 6, d = col & 63;
#pragma unroll
      for (int r = 0; r < 8; ++r) {
        int row = row0 + 8 * lh + r;
        int bb = row & 7, j = row >> 3;
        out0[(((size_t)bb * N_H + nh) * Q_LEN + j) * H_D + d] = (__bf16)c[r];
      }
    } else if (mode == 2) {                // V -> transposed [b][n][d][j]
      int nh = col >> 6, d = col & 63;
#pragma unroll
      for (int r = 0; r < 8; ++r) {
        int row = row0 + 8 * lh + r;
        int bb = row & 7, j = row >> 3;
        out0[(((size_t)bb * N_H + nh) * H_D + d) * Q_LEN + j] = (__bf16)c[r];
      }
    } else if (mode == 3) {                // KR -> [b][n][jr][d]
      int nh = col >> 6, d = col & 63;
#pragma unroll
      for (int r = 0; r < 8; ++r) {
        int row = row0 + 8 * lh + r;
        int bb = row & 7, jr = row >> 3;
        out0[(((size_t)bb * N_H + nh) * P_LEN + jr) * H_D + d] = (__bf16)c[r];
      }
    } else {                               // f32 row-major (output projection)
#pragma unroll
      for (int r = 0; r < 8; ++r) {
        int row = row0 + 8 * lh + r;
        outf[(size_t)row * N + col] = c[r];
      }
    }
  }
}

// ---------------- ef[i,b,n,s] = sum_d (Q+r_s_bias)[i,b,n,d]*seg_mat[s,n,d] --
// reconstructed from qw (= Q + r_w_bias):  (qw - r_w + r_s) . seg
__global__ __launch_bounds__(256) void ef_kernel(
    const __bf16* __restrict__ qw, const float* __restrict__ seg_mat,
    const float* __restrict__ rwb, const float* __restrict__ rsb,
    float* __restrict__ ef) {
  int idx = blockIdx.x * 256 + threadIdx.x;     // 4096*16*2 = 131072
  int s = idx & 1, n = (idx >> 1) & 15, m = idx >> 5;
  const __bf16* qp = qw + (size_t)m * HIDN + n * H_D;
  const float* sp = seg_mat + ((size_t)s * N_H + n) * H_D;
  const float* wb = rwb + n * H_D;
  const float* sb = rsb + n * H_D;
  float acc = 0.f;
#pragma unroll 8
  for (int d = 0; d < H_D; ++d)
    acc += ((float)qp[d] - wb[d] + sb[d]) * sp[d];
  ef[idx] = acc;
}

// ---------------- fused flash attention (one 16-row i-tile per wave) -------
// S^T formulation: each lane owns one query column i -> softmax stats are
// per-lane scalars; P^T B-frags are assembled with one shfl_xor(16).
// rel_shift(bd, klen)[i,j] == bd_raw[i, j - i + 512]; per 32-j chunk this
// needs a 48-row diagonal band of q_r . kr^T, computed with 3 WMMA tiles
// and gathered through a wave-private LDS buffer.
__global__ __launch_bounds__(256) void attn_kernel(
    const __bf16* __restrict__ qw, const __bf16* __restrict__ qr,
    const __bf16* __restrict__ kbuf, const __bf16* __restrict__ vt,
    const __bf16* __restrict__ krbuf, const float* __restrict__ efb,
    const float* __restrict__ seg, const float* __restrict__ mask,
    __bf16* __restrict__ attn_out) {
  __shared__ float band[8][48 * 16];         // 24 KB: 48-row bd band per wave
  int wave = threadIdx.x >> 5, lane = threadIdx.x & 31;
  int li = lane & 15, lh = lane >> 4;
  int task = blockIdx.x * 8 + wave;          // 4096 tasks total
  int i0 = (task & 31) * 16;
  int n  = (task >> 5) & 15;
  int b  = task >> 9;
  int i  = i0 + li;                          // query column owned by lane

  const __bf16* qw_row = qw + ((size_t)i * B_SZ + b) * HIDN + n * H_D;
  const __bf16* qr_row = qr + ((size_t)i * B_SZ + b) * HIDN + n * H_D;
  const __bf16* kb  = kbuf  + ((size_t)b * N_H + n) * Q_LEN * H_D;
  const __bf16* krb = krbuf + ((size_t)b * N_H + n) * P_LEN * H_D;
  const __bf16* vb  = vt    + ((size_t)b * N_H + n) * H_D * Q_LEN;

  float ef0 = efb[((size_t)i * B_SZ + b) * N_H * 2 + n * 2 + 0];
  float ef1 = efb[((size_t)i * B_SZ + b) * N_H * 2 + n * 2 + 1];

  // loop-invariant Q^T B-fragments (d0 = 0 and 32)
  v16bf bqw0 = *(const v16bf*)(qw_row + 16 * lh);
  v16bf bqw1 = *(const v16bf*)(qw_row + 32 + 16 * lh);
  v16bf bqr0 = *(const v16bf*)(qr_row + 16 * lh);
  v16bf bqr1 = *(const v16bf*)(qr_row + 32 + 16 * lh);

  v8f o0 = {}, o1 = {}, o2 = {}, o3 = {};
  float m_run = -INFINITY, l_run = 0.f;
  float* bandw = band[wave];

  for (int j0 = 0; j0 < Q_LEN; j0 += 32) {
    // ---- bd diagonal band: rows jr in [jrb, jrb+48) ----
    int jrb = j0 - i0 + 496;                 // always in [0, 976]
#pragma unroll
    for (int t = 0; t < 3; ++t) {
      v8f f = {};
      f = wmma_bf16(load_a_frag(krb, H_D, jrb + t * 16, 0,  lane), bqr0, f);
      f = wmma_bf16(load_a_frag(krb, H_D, jrb + t * 16, 32, lane), bqr1, f);
#pragma unroll
      for (int r = 0; r < 8; ++r)
        bandw[(t * 16 + 8 * lh + r) * 16 + li] = f[r];
    }
    asm volatile("s_wait_dscnt 0" ::: "memory");   // wave-private LDS RAW

    // ---- S^T = ac + shift(bd) + ef, scaled, masked ----
    float sv[16];
#pragma unroll
    for (int t = 0; t < 2; ++t) {
      v8f s = {};
      s = wmma_bf16(load_a_frag(kb, H_D, j0 + t * 16, 0,  lane), bqw0, s);
      s = wmma_bf16(load_a_frag(kb, H_D, j0 + t * 16, 32, lane), bqw1, s);
#pragma unroll
      for (int r = 0; r < 8; ++r) {
        int jrow = t * 16 + 8 * lh + r;      // j - j0
        int j = j0 + jrow;
        float bd = bandw[(16 + jrow - li) * 16 + li];   // jr = j - i + 512
        const float* sp = seg + (((size_t)i * Q_LEN + j) * B_SZ + b) * 2;
        float efv = ef0 * sp[0] + ef1 * sp[1];
        float mk = mask[((size_t)i * Q_LEN + j) * B_SZ + b];
        sv[t * 8 + r] = (s[r] + bd + efv) * SCALE_F - BIG_F * mk;
      }
    }

    // ---- online softmax over the 32-j chunk (stats per column i) ----
    float cmax = sv[0];
#pragma unroll
    for (int e = 1; e < 16; ++e) cmax = fmaxf(cmax, sv[e]);
    cmax = fmaxf(cmax, __shfl_xor(cmax, 16, 32));
    float m_new = fmaxf(m_run, cmax);
    float corr = __expf(m_run - m_new);
    float p[16], psum = 0.f;
#pragma unroll
    for (int e = 0; e < 16; ++e) { p[e] = __expf(sv[e] - m_new); psum += p[e]; }
    psum += __shfl_xor(psum, 16, 32);
    l_run = l_run * corr + psum;
    m_run = m_new;
#pragma unroll
    for (int r = 0; r < 8; ++r) {
      o0[r] *= corr; o1[r] *= corr; o2[r] *= corr; o3[r] *= corr;
    }

    // ---- assemble P^T B-fragment (32j x 16i) via shfl_xor(16) ----
    v16bf bp;
#pragma unroll
    for (int r = 0; r < 8; ++r) {
      float x0 = __shfl_xor(p[r],     16, 32);
      float x1 = __shfl_xor(p[8 + r], 16, 32);
      bp[r]     = (__bf16)(lh ? x1 : p[r]);
      bp[8 + r] = (__bf16)(lh ? p[8 + r] : x0);
    }

    // ---- O^T += V^T . P^T  (4 d-tiles, k = 32 j's) ----
    o0 = wmma_bf16(load_a_frag(vb, Q_LEN, 0,  j0, lane), bp, o0);
    o1 = wmma_bf16(load_a_frag(vb, Q_LEN, 16, j0, lane), bp, o1);
    o2 = wmma_bf16(load_a_frag(vb, Q_LEN, 32, j0, lane), bp, o2);
    o3 = wmma_bf16(load_a_frag(vb, Q_LEN, 48, j0, lane), bp, o3);
  }

  // ---- normalize and scatter attn_out[i*8+b][n*64+d] as bf16 ----
  float inv = 1.f / l_run;
  __bf16* op = attn_out + ((size_t)i * B_SZ + b) * HIDN + n * H_D;
#pragma unroll
  for (int r = 0; r < 8; ++r) {
    int dr = 8 * lh + r;
    op[dr]      = (__bf16)(o0[r] * inv);
    op[16 + dr] = (__bf16)(o1[r] * inv);
    op[32 + dr] = (__bf16)(o2[r] * inv);
    op[48 + dr] = (__bf16)(o3[r] * inv);
  }
}

// ---------------- residual + layernorm, one row per block -----------------
__global__ __launch_bounds__(256) void ln_kernel(
    const float* __restrict__ proj, const float* __restrict__ hidden,
    const float* __restrict__ gamma, const float* __restrict__ beta,
    float* __restrict__ out) {
  __shared__ float red[8];
  int m = blockIdx.x, tid = threadIdx.x;
  const float* xp = proj + (size_t)m * HIDN;
  const float* hp = hidden + (size_t)m * HIDN;
  float vals[4], s = 0.f;
#pragma unroll
  for (int e = 0; e < 4; ++e) { vals[e] = xp[tid + e * 256] + hp[tid + e * 256]; s += vals[e]; }
#pragma unroll
  for (int off = 16; off >= 1; off >>= 1) s += __shfl_xor(s, off, 32);
  if ((tid & 31) == 0) red[tid >> 5] = s;
  __syncthreads();
  float tot = 0.f;
#pragma unroll
  for (int w = 0; w < 8; ++w) tot += red[w];
  float mu = tot * (1.f / HIDN);
  __syncthreads();
  float v = 0.f;
#pragma unroll
  for (int e = 0; e < 4; ++e) { float d = vals[e] - mu; v += d * d; }
#pragma unroll
  for (int off = 16; off >= 1; off >>= 1) v += __shfl_xor(v, off, 32);
  if ((tid & 31) == 0) red[tid >> 5] = v;
  __syncthreads();
  float vt = 0.f;
#pragma unroll
  for (int w = 0; w < 8; ++w) vt += red[w];
  float rstd = rsqrtf(vt * (1.f / HIDN) + 1e-5f);
#pragma unroll
  for (int e = 0; e < 4; ++e) {
    int h = tid + e * 256;
    out[(size_t)m * HIDN + h] = (vals[e] - mu) * rstd * gamma[h] + beta[h];
  }
}

// ---------------- workspace layout (all 256B-aligned) ----------------------
constexpr size_t OFF_HID  = 0;                                  // 4096x1024 bf16
constexpr size_t OFF_POS  = OFF_HID  + (size_t)MROWS * HIDN * 2;
constexpr size_t OFF_WQT  = OFF_POS  + (size_t)PROWS * HIDN * 2;
constexpr size_t OFF_WKT  = OFF_WQT  + (size_t)HIDN * HIDN * 2;
constexpr size_t OFF_WVT  = OFF_WKT  + (size_t)HIDN * HIDN * 2;
constexpr size_t OFF_WRT  = OFF_WVT  + (size_t)HIDN * HIDN * 2;
constexpr size_t OFF_WO   = OFF_WRT  + (size_t)HIDN * HIDN * 2;
constexpr size_t OFF_QW   = OFF_WO   + (size_t)HIDN * HIDN * 2;
constexpr size_t OFF_QR   = OFF_QW   + (size_t)MROWS * HIDN * 2;
constexpr size_t OFF_KB   = OFF_QR   + (size_t)MROWS * HIDN * 2;
constexpr size_t OFF_VT   = OFF_KB   + (size_t)MROWS * HIDN * 2;
constexpr size_t OFF_KR   = OFF_VT   + (size_t)MROWS * HIDN * 2;
constexpr size_t OFF_EF   = OFF_KR   + (size_t)PROWS * HIDN * 2;
constexpr size_t OFF_AOUT = OFF_EF   + (size_t)MROWS * N_H * 2 * 4;
constexpr size_t OFF_PROJ = OFF_AOUT + (size_t)MROWS * HIDN * 2;

extern "C" void kernel_launch(void* const* d_in, const int* in_sizes, int n_in,
                              void* d_out, int out_size, void* d_ws, size_t ws_size,
                              hipStream_t stream) {
  (void)in_sizes; (void)n_in; (void)out_size; (void)ws_size;
  const float* hidden = (const float*)d_in[0];
  const float* posemb = (const float*)d_in[1];
  const float* segemb = (const float*)d_in[2];
  const float* amask  = (const float*)d_in[3];
  const float* wq = (const float*)d_in[4];
  const float* wk = (const float*)d_in[5];
  const float* wv = (const float*)d_in[6];
  const float* wr = (const float*)d_in[7];
  const float* wo = (const float*)d_in[8];
  const float* rwb = (const float*)d_in[9];
  const float* rrb = (const float*)d_in[10];
  const float* rsb = (const float*)d_in[11];
  const float* segm = (const float*)d_in[12];
  const float* lng = (const float*)d_in[13];
  const float* lnb = (const float*)d_in[14];
  float* out = (float*)d_out;

  char* ws = (char*)d_ws;
  __bf16* hid_bf = (__bf16*)(ws + OFF_HID);
  __bf16* pos_bf = (__bf16*)(ws + OFF_POS);
  __bf16* wqT = (__bf16*)(ws + OFF_WQT);
  __bf16* wkT = (__bf16*)(ws + OFF_WKT);
  __bf16* wvT = (__bf16*)(ws + OFF_WVT);
  __bf16* wrT = (__bf16*)(ws + OFF_WRT);
  __bf16* woB = (__bf16*)(ws + OFF_WO);
  __bf16* qwB = (__bf16*)(ws + OFF_QW);
  __bf16* qrB = (__bf16*)(ws + OFF_QR);
  __bf16* kB  = (__bf16*)(ws + OFF_KB);
  __bf16* vT  = (__bf16*)(ws + OFF_VT);
  __bf16* krB = (__bf16*)(ws + OFF_KR);
  float*  efB = (float*)(ws + OFF_EF);
  __bf16* aout = (__bf16*)(ws + OFF_AOUT);
  float*  proj = (float*)(ws + OFF_PROJ);

  // 1) precision / layout prep
  cvt_bf16_kernel<<<(MROWS * HIDN) / 256, 256, 0, stream>>>(hidden, hid_bf, (size_t)MROWS * HIDN);
  cvt_bf16_kernel<<<(PROWS * HIDN) / 256, 256, 0, stream>>>(posemb, pos_bf, (size_t)PROWS * HIDN);
  dim3 tg(HIDN / 16, HIDN / 16);
  wtrans_kernel<<<tg, 256, 0, stream>>>(wq, wqT);
  wtrans_kernel<<<tg, 256, 0, stream>>>(wk, wkT);
  wtrans_kernel<<<tg, 256, 0, stream>>>(wv, wvT);
  wtrans_kernel<<<tg, 256, 0, stream>>>(wr, wrT);
  cvt_bf16_kernel<<<(HIDN * HIDN) / 256, 256, 0, stream>>>(wo, woB, (size_t)HIDN * HIDN);

  // 2) projections (WMMA) -- waves = (M/16)*(N/64), 8 waves per block
  int g_m = (MROWS / 16) * (HIDN / 64) / 8;   // 512 blocks
  int g_p = (PROWS / 16) * (HIDN / 64) / 8;   // 1024 blocks
  gemm_kernel<<<g_m, 256, 0, stream>>>(hid_bf, wqT, HIDN / 64, HIDN, 0, rwb, rrb, qwB, qrB, nullptr);
  gemm_kernel<<<g_m, 256, 0, stream>>>(hid_bf, wkT, HIDN / 64, HIDN, 1, nullptr, nullptr, kB, nullptr, nullptr);
  gemm_kernel<<<g_m, 256, 0, stream>>>(hid_bf, wvT, HIDN / 64, HIDN, 2, nullptr, nullptr, vT, nullptr, nullptr);
  gemm_kernel<<<g_p, 256, 0, stream>>>(pos_bf, wrT, HIDN / 64, HIDN, 3, nullptr, nullptr, krB, nullptr, nullptr);

  // 3) segment-bias dot products
  ef_kernel<<<(MROWS * N_H * 2) / 256, 256, 0, stream>>>(qwB, segm, rwb, rsb, efB);

  // 4) fused attention: ac + rel-shifted bd + ef + mask + softmax + PV (WMMA)
  //    tasks = 32 i-tiles * 16 heads * 8 batch = 4096 waves = 512 blocks
  attn_kernel<<<512, 256, 0, stream>>>(
      qwB, qrB, kB, vT, krB, efB, segemb, amask, aout);

  // 5) output projection (WMMA) then residual + layernorm
  gemm_kernel<<<g_m, 256, 0, stream>>>(aout, woB, HIDN / 64, HIDN, 4, nullptr, nullptr, nullptr, nullptr, proj);
  ln_kernel<<<MROWS, 256, 0, stream>>>(proj, hidden, lng, lnb, out);
}
